// MambaMixer_51522427683044
// MI455X (gfx1250) — matrix operand order repaired and checked by reading, not compiled, not run on hardware
//
#include <hip/hip_runtime.h>

// ---------------------------------------------------------------------------
// Types
// ---------------------------------------------------------------------------
typedef __attribute__((ext_vector_type(16))) __bf16        v16bf;
typedef __attribute__((ext_vector_type(8)))  float         v8f;
typedef __attribute__((ext_vector_type(4)))  unsigned int  v4u;
typedef __attribute__((ext_vector_type(8)))  int           v8i;
typedef __attribute__((ext_vector_type(4)))  int           v4i;

union AFrag { unsigned int u[8]; v16bf v; };

__device__ __forceinline__ unsigned short f32_to_bf16(float f) {
    unsigned int u = __float_as_uint(f);
    unsigned int r = u + 0x7FFFu + ((u >> 16) & 1u);   // round-to-nearest-even
    return (unsigned short)(r >> 16);
}

// ---------------------------------------------------------------------------
// Tensor Data Mover availability
// ---------------------------------------------------------------------------
#if defined(__has_builtin)
#if __has_builtin(__builtin_amdgcn_tensor_load_to_lds) && \
    __has_builtin(__builtin_amdgcn_s_wait_tensorcnt)
#define USE_TDM 1
#endif
#endif
#ifndef USE_TDM
#define USE_TDM 0
#endif

#if USE_TDM
// 2-D bf16 tile load, global -> LDS, via TDM descriptor (ISA 8.3/8.4).
// dim0/stride0 in elements (2B); tile0 x tile1 tile starting at gptr.
// 6-arg builtin form (clang-23 / therock-10.0): (v4u, v8i, v4i, v4i, v8i, i32)
__device__ __forceinline__ void tdm_load_2d(unsigned lds_addr,
                                            const unsigned short* gptr,
                                            int dim0, int dim1, int stride0,
                                            int tile0, int tile1) {
    unsigned long long ga = (unsigned long long)(size_t)gptr;
    v4u g0;
    g0[0] = 1u;                                            // count=1
    g0[1] = lds_addr;                                      // lds_addr
    g0[2] = (unsigned)(ga & 0xFFFFFFFFull);                // global_addr[31:0]
    g0[3] = (unsigned)((ga >> 32) & 0x01FFFFFFull)         // global_addr[56:32]
          | 0x80000000u;                                   // type=2 (image)
    v8i g1;
    g1[0] = (int)(1u << 16);                               // data_size=1 (2 bytes)
    g1[1] = (int)(((unsigned)dim0 & 0xFFFFu) << 16);       // tensor_dim0[15:0]
    g1[2] = (int)((((unsigned)dim0 >> 16) & 0xFFFFu)
          | (((unsigned)dim1 & 0xFFFFu) << 16));           // dim0[31:16]|dim1[15:0]
    g1[3] = (int)((((unsigned)dim1 >> 16) & 0xFFFFu)
          | (((unsigned)tile0 & 0xFFFFu) << 16));          // dim1[31:16]|tile_dim0
    g1[4] = (int)((unsigned)tile1 & 0xFFFFu);              // tile_dim1 (tile_dim2=0)
    g1[5] = (int)(unsigned)stride0;                        // tensor_dim0_stride[31:0]
    g1[6] = 0;
    g1[7] = 0;
    v4i z4 = {0, 0, 0, 0};
    v8i z8 = {0, 0, 0, 0, 0, 0, 0, 0};
    __builtin_amdgcn_tensor_load_to_lds(g0, g1, z4, z4, z8, 0);
}
#endif

// ---------------------------------------------------------------------------
// WMMA GEMM: C[M,N] = A[M,K] @ B[K,N]; A bf16 [M,K] (row stride lda),
// Bt bf16 = B transposed [N,K]. 256 thr = 8 waves; block tile 256(M) x 32(N);
// wave tile 64x16 -> 4 x v_wmma_f32_16x16x32_bf16 per K-chunk (B frag reused).
// epi: 0=none, 1=softplus(x + bias[col]). Cbf: optional bf16 shadow of C.
// ---------------------------------------------------------------------------
#define BLK_M 256
#define BLK_N 32
#define BLK_K 32

__global__ __launch_bounds__(256)
void gemm_wmma_bf16(const unsigned short* __restrict__ A,
                    const unsigned short* __restrict__ Bt,
                    float* __restrict__ C, unsigned short* __restrict__ Cbf,
                    const float* __restrict__ bias,
                    int M, int N, int K, int lda, int epi) {
    __shared__ __align__(16) unsigned short sA[BLK_M * BLK_K];  // [m][k]
    __shared__ __align__(16) unsigned short sB[BLK_N * BLK_K];  // [n][k]

    const int tid  = threadIdx.x;
    const int lane = tid & 31;
    const int wave = tid >> 5;
    const int wM   = wave >> 1;          // 0..3 -> 64-row group
    const int wN   = wave & 1;           // 0..1 -> 16-col group
    const int hi   = lane >> 4;
    const int l15  = lane & 15;
    const int m0   = blockIdx.x * BLK_M;
    const int n0   = blockIdx.y * BLK_N;

    v8f acc[4] = {};

#if USE_TDM
    const unsigned ldsA = (unsigned)(size_t)(void*)sA;
    const unsigned ldsB = (unsigned)(size_t)(void*)sB;
#endif

    for (int k0 = 0; k0 < K; k0 += BLK_K) {
#if USE_TDM
        if (wave == 0) {
            tdm_load_2d(ldsA, A + (size_t)m0 * lda + k0, lda, M, lda, BLK_K, BLK_M);
            tdm_load_2d(ldsB, Bt + (size_t)n0 * K + k0, K, N, K, BLK_K, BLK_N);
            __builtin_amdgcn_s_wait_tensorcnt(0);
        }
#else
        for (int i = tid; i < (BLK_M * BLK_K) / 8; i += 256) {  // uint4 = 8 bf16
            int r = i >> 2, q = i & 3;
            *(uint4*)&sA[r * BLK_K + q * 8] =
                *(const uint4*)&A[(size_t)(m0 + r) * lda + k0 + q * 8];
        }
        if (tid < (BLK_N * BLK_K) / 8) {
            int r = tid >> 2, q = tid & 3;
            *(uint4*)&sB[r * BLK_K + q * 8] =
                *(const uint4*)&Bt[(size_t)(n0 + r) * K + k0 + q * 8];
        }
#endif
        __syncthreads();

        // B fragment: lane holds col N=l15; lanes 16-31 take K=16..31
        AFrag bfr;
        const int bbase = (wN * 16 + l15) * BLK_K + (hi << 4);
        #pragma unroll
        for (int v = 0; v < 8; ++v)
            bfr.u[v] = *(const unsigned int*)&sB[bbase + (v << 1)];

        #pragma unroll
        for (int s = 0; s < 4; ++s) {
            // A fragment: lane row M=l15; lanes 16-31 take K=16..31
            AFrag afr;
            const int abase = (wM * 64 + s * 16 + l15) * BLK_K;
            #pragma unroll
            for (int v = 0; v < 8; ++v) {
                int kk = (hi << 4) + (v << 1);
                afr.u[v] = *(const unsigned int*)&sA[abase + kk];
            }
            acc[s] = __builtin_amdgcn_wmma_f32_16x16x32_bf16(
                false, afr.v, false, bfr.v, (short)0, acc[s], false, false);
        }
        __syncthreads();
    }

    const int col = n0 + wN * 16 + l15;
    const float bcol = (epi == 1) ? bias[col] : 0.0f;
    #pragma unroll
    for (int s = 0; s < 4; ++s) {
        const int rowbase = m0 + wM * 64 + s * 16 + hi * 8;
        #pragma unroll
        for (int r = 0; r < 8; ++r) {
            float v = acc[s][r];
            if (epi == 1) {
                v += bcol;
                float sp = __logf(1.0f + __expf(v));   // cheap softplus
                v = (v > 20.0f) ? v : sp;
            }
            size_t off = (size_t)(rowbase + r) * N + col;
            C[off] = v;
            if (Cbf) Cbf[off] = f32_to_bf16(v);
        }
    }
}

// ---------------------------------------------------------------------------
// f32 -> bf16 elementwise
// ---------------------------------------------------------------------------
__global__ __launch_bounds__(256)
void convert_bf16(const float* __restrict__ in, unsigned short* __restrict__ out) {
    int i = blockIdx.x * 256 + threadIdx.x;
    out[i] = f32_to_bf16(in[i]);
}

// ---------------------------------------------------------------------------
// [K,N] f32 -> [N,K] bf16 transpose (32x32 LDS tile)
// ---------------------------------------------------------------------------
__global__ __launch_bounds__(256)
void transpose_to_bf16(const float* __restrict__ in,
                       unsigned short* __restrict__ out, int K, int N) {
    __shared__ float tile[32][33];
    const int k0 = blockIdx.x * 32, n0 = blockIdx.y * 32;
    const int tx = threadIdx.x & 31, ty = threadIdx.x >> 5;
    for (int r = ty; r < 32; r += 8)
        tile[r][tx] = in[(size_t)(k0 + r) * N + n0 + tx];
    __syncthreads();
    for (int r = ty; r < 32; r += 8)
        out[(size_t)(n0 + r) * K + k0 + tx] = f32_to_bf16(tile[tx][r]);
}

// ---------------------------------------------------------------------------
// Causal depthwise conv1d (D_CONV=4) + SiLU; writes f32 (for scan) + bf16 (A of
// next GEMM). x_in = first 2048 cols of xz (row stride 4096).
// ---------------------------------------------------------------------------
__global__ __launch_bounds__(256)
void conv_silu_kernel(const float* __restrict__ xz,
                      const float* __restrict__ conv_w,
                      const float* __restrict__ conv_b,
                      float* __restrict__ xc,
                      unsigned short* __restrict__ xc_bf) {
    int idx = blockIdx.x * 256 + threadIdx.x;
    int t = idx >> 11;
    int d = idx & 2047;
    float acc = conv_b[d];
    #pragma unroll
    for (int j = 0; j < 4; ++j) {
        int ts = t - 3 + j;
        if (ts >= 0) acc += xz[(size_t)ts * 4096 + d] * conv_w[d * 4 + j];
    }
    float v = acc / (1.0f + __expf(-acc));             // SiLU
    xc[idx] = v;
    xc_bf[idx] = f32_to_bf16(v);
}

// ---------------------------------------------------------------------------
// Selective scan: one channel per lane (2048 ch = 8 blocks x 256 thr), h[16] in
// VGPRs, B_t/C_t staged in LDS per 32-step chunk. Fuses +xc*Dp and *silu(z);
// writes bf16 y (A-matrix of the output GEMM).
// ---------------------------------------------------------------------------
__global__ __launch_bounds__(256)
void scan_kernel(const float* __restrict__ dt,
                 const float* __restrict__ x_dbl,   // [L,96]: B=64..79, C=80..95
                 const float* __restrict__ xc,
                 const float* __restrict__ xz,      // z = cols 2048..4095 (stride 4096)
                 const float* __restrict__ A_log,
                 const float* __restrict__ Dp,
                 unsigned short* __restrict__ y_bf) {
    __shared__ float sBC[32 * 32];

    const int tid = threadIdx.x;
    const int d   = blockIdx.x * 256 + tid;

    float Ar[16], h[16];
    #pragma unroll
    for (int n = 0; n < 16; ++n) {
        Ar[n] = -__expf(A_log[d * 16 + n]);
        h[n]  = 0.0f;
    }
    const float dp = Dp[d];

    for (int t0 = 0; t0 < 2048; t0 += 32) {
        for (int i = tid; i < 1024; i += 256) {
            int tl = i >> 5, c = i & 31;
            sBC[i] = x_dbl[(size_t)(t0 + tl) * 96 + 64 + c];
        }
        __syncthreads();

        for (int tl = 0; tl < 32; ++tl) {
            const int t = t0 + tl;
            const float dtv = dt[(size_t)t * 2048 + d];
            const float xv  = xc[(size_t)t * 2048 + d];
            const float* Bv = &sBC[tl * 32];
            const float* Cv = Bv + 16;
            float accy = 0.0f;
            #pragma unroll
            for (int n = 0; n < 16; ++n) {
                float dA = __expf(dtv * Ar[n]);
                h[n] = dA * h[n] + (dtv * Bv[n]) * xv;
                accy += h[n] * Cv[n];
            }
            const float zv = xz[(size_t)t * 4096 + 2048 + d];
            const float g  = zv / (1.0f + __expf(-zv));
            y_bf[(size_t)t * 2048 + d] = f32_to_bf16((accy + xv * dp) * g);
        }
        __syncthreads();
    }
}

// ---------------------------------------------------------------------------
// Launch
// ---------------------------------------------------------------------------
extern "C" void kernel_launch(void* const* d_in, const int* in_sizes, int n_in,
                              void* d_out, int out_size, void* d_ws, size_t ws_size,
                              hipStream_t stream) {
    const float* x      = (const float*)d_in[0];   // [2048,1024]
    const float* w_in   = (const float*)d_in[1];   // [1024,4096]
    const float* conv_w = (const float*)d_in[2];   // [2048,1,4]
    const float* conv_b = (const float*)d_in[3];   // [2048]
    const float* w_xprj = (const float*)d_in[4];   // [2048,96]
    const float* w_dt   = (const float*)d_in[5];   // [64,2048]
    const float* b_dt   = (const float*)d_in[6];   // [2048]
    const float* A_log  = (const float*)d_in[7];   // [2048,16]
    const float* Dp     = (const float*)d_in[8];   // [2048]
    const float* w_out  = (const float*)d_in[9];   // [2048,1024]
    float* out = (float*)d_out;                    // [2048,1024]

    char* ws = (char*)d_ws;
    float*          xz      = (float*)(ws);                       // 32 MB [2048,4096]
    float*          xc      = (float*)(ws + (size_t)33554432);    // 16 MB [2048,2048]
    float*          x_dbl   = (float*)(ws + (size_t)50331648);    // [2048,96]
    float*          dt      = (float*)(ws + (size_t)51118080);    // 16 MB [2048,2048]
    unsigned short* x_bf    = (unsigned short*)(ws + (size_t)67895296);  // [2048,1024]
    unsigned short* w_in_t  = (unsigned short*)(ws + (size_t)72089600);  // [4096,1024]
    unsigned short* w_xp_t  = (unsigned short*)(ws + (size_t)80478208);  // [96,2048]
    unsigned short* w_dt_t  = (unsigned short*)(ws + (size_t)80871424);  // [2048,64]
    unsigned short* w_out_t = (unsigned short*)(ws + (size_t)81133568);  // [1024,2048]
    unsigned short* xc_bf   = (unsigned short*)(ws + (size_t)85327872);  // [2048,2048]
    unsigned short* xdbl_bf = (unsigned short*)(ws + (size_t)93716480);  // [2048,96]
    unsigned short* y_bf    = (unsigned short*)(ws + (size_t)94109696);  // [2048,2048]

    const int L = 2048, DI = 2048, DM = 1024;

    // precision staging: activations/weights -> bf16 (weights transposed to [N,K])
    convert_bf16<<<(L * DM) / 256, 256, 0, stream>>>(x, x_bf);
    transpose_to_bf16<<<dim3(DM / 32, 4096 / 32), 256, 0, stream>>>(w_in, w_in_t, DM, 4096);
    transpose_to_bf16<<<dim3(DI / 32, 96 / 32), 256, 0, stream>>>(w_xprj, w_xp_t, DI, 96);
    transpose_to_bf16<<<dim3(64 / 32, DI / 32), 256, 0, stream>>>(w_dt, w_dt_t, 64, DI);
    transpose_to_bf16<<<dim3(DI / 32, DM / 32), 256, 0, stream>>>(w_out, w_out_t, DI, DM);

    // 1) xz = x @ w_in
    gemm_wmma_bf16<<<dim3(L / BLK_M, 4096 / BLK_N), 256, 0, stream>>>(
        x_bf, w_in_t, xz, nullptr, nullptr, L, 2 * DI, DM, DM, 0);

    // 2) causal conv + SiLU
    conv_silu_kernel<<<(L * DI) / 256, 256, 0, stream>>>(xz, conv_w, conv_b, xc, xc_bf);

    // 3) x_dbl = xc @ w_xproj (+ bf16 shadow for the dt GEMM)
    gemm_wmma_bf16<<<dim3(L / BLK_M, 96 / BLK_N), 256, 0, stream>>>(
        xc_bf, w_xp_t, x_dbl, xdbl_bf, nullptr, L, 96, DI, DI, 0);

    // 4) dt = softplus(x_dbl[:, :64] @ w_dt + b_dt)   (A view: lda=96)
    gemm_wmma_bf16<<<dim3(L / BLK_M, DI / BLK_N), 256, 0, stream>>>(
        xdbl_bf, w_dt_t, dt, nullptr, b_dt, L, DI, 64, 96, 1);

    // 5) selective scan (fused Dp skip + silu(z) gating) -> y_bf
    scan_kernel<<<DI / 256, 256, 0, stream>>>(dt, x_dbl, xc, xz, A_log, Dp, y_bf);

    // 6) out = y @ w_out
    gemm_wmma_bf16<<<dim3(L / BLK_M, DM / BLK_N), 256, 0, stream>>>(
        y_bf, w_out_t, out, nullptr, nullptr, L, DM, DI, DI, 0);
}
